// GAT2_43559558316090
// MI455X (gfx1250) — compile-verified, matrix operand order checked by reading, not run on hardware
//
#include <hip/hip_runtime.h>
#include <hip/hip_bf16.h>

// ---------------------------------------------------------------------------
// GAT x4 + mean-pool + MLP for MI455X (gfx1250, wave32).
// GEMMs: v_wmma_f32_16x16x32_bf16 (f32 accumulate), 4 WMMA tiles per wave with
// A-operand register reuse. Activations/weights are pre-cast to bf16 so the
// GEMM inner loop is pure global_load_b128 + WMMA (no VALU -> no hazard NOPs).
// Edge softmax/aggregate: f32 global atomics (per-layer h <= 102MB fits the
// 192MB L2 -> random gathers are L2 hits).
// ---------------------------------------------------------------------------

typedef __attribute__((ext_vector_type(16))) __bf16 v16bf;
typedef __attribute__((ext_vector_type(8)))  __bf16 v8bf;
typedef __attribute__((ext_vector_type(8)))  float  v8f;

#define NEG_SLOPE 0.2f
#define HEADS 8
#define NUM_GRAPHS 64

// 16 K-values for this lane's (row, half) slot per the 16-bit A/B VGPR layout:
// elements 0..7 = K base..base+7, elements 8..15 = K base+16..base+23, where
// base = 8*half. Two 16B loads; shufflevector is pure register placement.
__device__ __forceinline__ v16bf load_bf32(const __bf16* __restrict__ p) {
    v8bf lo = *(const v8bf*)(p);
    v8bf hi = *(const v8bf*)(p + 16);
    return __builtin_shufflevector(lo, hi, 0, 1, 2, 3, 4, 5, 6, 7,
                                   8, 9, 10, 11, 12, 13, 14, 15);
}

// ---------------- GEMM: D[N,Fout] = A[N,K] @ W[Fout,K]^T  (bf16 WMMA) -------
// Block = 64 threads = 2 waves. Each wave computes a 16x64 strip of D
// (4 WMMA accumulators), reusing the A tile registers across 4 WMMAs.
// Requires N % 16 == 0 (N = 50000 = 16*3125), K % 32 == 0, Fout % 128 == 0.
__global__ void gemm_wmma_bf16(const __bf16* __restrict__ A,
                               const __bf16* __restrict__ W,
                               float* __restrict__ D,
                               int K, int Fout) {
    const int lane = threadIdx.x & 31;
    const int wave = threadIdx.x >> 5;                 // 0..1
    const int m0 = blockIdx.x * 16;
    const int n0 = (blockIdx.y * 2 + wave) * 64;

    const int half = lane >> 4;                        // K sub-slot
    const int l16  = lane & 15;

    const __bf16* Ap = A + (size_t)(m0 + l16) * K + half * 8;
    const __bf16* Wp = W + (size_t)(n0 + l16) * K + half * 8;
    const size_t wstride = (size_t)16 * K;             // 16 W rows per col tile

    v8f acc0 = {}, acc1 = {}, acc2 = {}, acc3 = {};
    for (int k0 = 0; k0 < K; k0 += 32) {
        v16bf av = load_bf32(Ap + k0);
        v16bf b0 = load_bf32(Wp + k0);
        v16bf b1 = load_bf32(Wp + wstride + k0);
        v16bf b2 = load_bf32(Wp + 2 * wstride + k0);
        v16bf b3 = load_bf32(Wp + 3 * wstride + k0);
        acc0 = __builtin_amdgcn_wmma_f32_16x16x32_bf16(false, av, false, b0, (short)0, acc0, false, false);
        acc1 = __builtin_amdgcn_wmma_f32_16x16x32_bf16(false, av, false, b1, (short)0, acc1, false, false);
        acc2 = __builtin_amdgcn_wmma_f32_16x16x32_bf16(false, av, false, b2, (short)0, acc2, false, false);
        acc3 = __builtin_amdgcn_wmma_f32_16x16x32_bf16(false, av, false, b3, (short)0, acc3, false, false);
    }

    // C/D layout: VGPR r -> row m0 + r + 8*half, col n0 + l16 (+16 per tile)
    float* Dp = D + (size_t)(m0 + half * 8) * Fout + n0 + l16;
#pragma unroll
    for (int r = 0; r < 8; ++r) {
        float* row = Dp + (size_t)r * Fout;
        row[0]  = acc0[r];
        row[16] = acc1[r];
        row[32] = acc2[r];
        row[48] = acc3[r];
    }
}

// ---------------- f32 -> bf16 cast (pre-pass for GEMM operands) -------------
__global__ void cast_bf16(const float* __restrict__ in, __bf16* __restrict__ out,
                          long n) {
    long i = (blockIdx.x * (long)blockDim.x + threadIdx.x) * 2;
    if (i + 1 < n) {
        float2 v = *(const float2*)(in + i);
        out[i]     = (__bf16)v.x;           // v_cvt_pk_bf16_f32 pair
        out[i + 1] = (__bf16)v.y;
    } else if (i < n) {
        out[i] = (__bf16)in[i];
    }
}

// ---------------- build edge list with self loops ---------------------------
__global__ void build_edges(const int* __restrict__ ei, int* __restrict__ srcE,
                            int* __restrict__ dstE, int E0, int N) {
    int i = blockIdx.x * blockDim.x + threadIdx.x;
    int Et = E0 + N;
    if (i >= Et) return;
    if (i < E0) { srcE[i] = ei[i]; dstE[i] = ei[E0 + i]; }
    else        { srcE[i] = i - E0; dstE[i] = i - E0; }
}

// ---------------- per-node per-head attention dot products ------------------
__global__ void attn_coef(const float* __restrict__ h, const float* __restrict__ as,
                          const float* __restrict__ ad, float* __restrict__ es,
                          float* __restrict__ ed, int N, int C) {
    int i = blockIdx.x * blockDim.x + threadIdx.x;      // n*H + head
    if (i >= N * HEADS) return;
    int hh = i & (HEADS - 1);
    int n  = i >> 3;
    const float* hp  = h  + (size_t)n * HEADS * C + hh * C;
    const float* asp = as + hh * C;
    const float* adp = ad + hh * C;
    float s = 0.f, d = 0.f;
    for (int c = 0; c < C; ++c) { float v = hp[c]; s += v * asp[c]; d += v * adp[c]; }
    es[i] = s; ed[i] = d;
}

// monotonic float <-> uint mapping for atomic max on signed floats
__device__ __forceinline__ unsigned f2ord(float f) {
    unsigned b = __float_as_uint(f);
    return (b & 0x80000000u) ? ~b : (b | 0x80000000u);
}
__device__ __forceinline__ float ord2f(unsigned u) {
    unsigned b = (u & 0x80000000u) ? (u & 0x7FFFFFFFu) : ~u;
    return __uint_as_float(b);
}
__device__ __forceinline__ float lrelu(float v) {
    return v > 0.f ? v : NEG_SLOPE * v;
}

// ---------------- edge pass A: segment max of logits ------------------------
__global__ void edge_max(const int* __restrict__ srcE, const int* __restrict__ dstE,
                         const float* __restrict__ es, const float* __restrict__ ed,
                         unsigned* __restrict__ mu, int Et) {
    int i = blockIdx.x * blockDim.x + threadIdx.x;      // e*H + head
    if (i >= Et * HEADS) return;
    int hh = i & (HEADS - 1);
    int e  = i >> 3;
    int s = srcE[e], d = dstE[e];
    float v = lrelu(es[s * HEADS + hh] + ed[d * HEADS + hh]);
    atomicMax(&mu[d * HEADS + hh], f2ord(v));
}

// ---------------- edge pass B: exp + segment sum ----------------------------
__global__ void edge_exp(const int* __restrict__ srcE, const int* __restrict__ dstE,
                         const float* __restrict__ es, const float* __restrict__ ed,
                         const unsigned* __restrict__ mu, float* __restrict__ exv,
                         float* __restrict__ denom, int Et) {
    int i = blockIdx.x * blockDim.x + threadIdx.x;
    if (i >= Et * HEADS) return;
    int hh = i & (HEADS - 1);
    int e  = i >> 3;
    int s = srcE[e], d = dstE[e];
    float v = lrelu(es[s * HEADS + hh] + ed[d * HEADS + hh]);
    float m = ord2f(mu[d * HEADS + hh]);
    float ex = expf(v - m);
    exv[i] = ex;
    atomicAdd(&denom[d * HEADS + hh], ex);
}

// ---------------- edge pass C: weighted scatter-aggregate -------------------
// grid.x = Et, block = Fout threads. agg[dst, f] += alpha(e,head(f)) * h[src, f]
__global__ void edge_scatter(const int* __restrict__ srcE, const int* __restrict__ dstE,
                             const float* __restrict__ hbuf, const float* __restrict__ exv,
                             const float* __restrict__ denom, float* __restrict__ agg,
                             int logC, int Fout) {
    int e = blockIdx.x;
    int t = threadIdx.x;
    int s = srcE[e], d = dstE[e];
    int hh = t >> logC;
    float alpha = exv[(size_t)e * HEADS + hh] / denom[d * HEADS + hh];
    atomicAdd(&agg[(size_t)d * Fout + t], alpha * hbuf[(size_t)s * Fout + t]);
}

// ---------------- bias + optional ELU ---------------------------------------
__global__ void bias_act(float* __restrict__ agg, const float* __restrict__ b,
                         long total, int Fout, int do_elu) {
    long i = blockIdx.x * (long)blockDim.x + threadIdx.x;
    if (i >= total) return;
    int f = (int)(i % Fout);
    float v = agg[i] + b[f];
    if (do_elu) v = v > 0.f ? v : (expf(v) - 1.f);
    agg[i] = v;
}

// ---------------- zero init -------------------------------------------------
__global__ void zero_f32(float* __restrict__ p, long n) {
    long i = blockIdx.x * (long)blockDim.x + threadIdx.x;
    if (i < n) p[i] = 0.f;
}

// ---------------- pooling + MLP ---------------------------------------------
__global__ void pool_sum(const float* __restrict__ h, const int* __restrict__ batch,
                         float* __restrict__ sums, int N, int F) {
    long i = blockIdx.x * (long)blockDim.x + threadIdx.x;
    if (i >= (long)N * F) return;
    int n = (int)(i / F), f = (int)(i % F);
    atomicAdd(&sums[batch[n] * F + f], h[i]);
}
__global__ void pool_cnt(const int* __restrict__ batch, float* __restrict__ cnt, int N) {
    int i = blockIdx.x * blockDim.x + threadIdx.x;
    if (i < N) atomicAdd(&cnt[batch[i]], 1.f);
}
__global__ void pool_div(float* __restrict__ sums, const float* __restrict__ cnt, int F) {
    int i = blockIdx.x * blockDim.x + threadIdx.x;
    if (i >= NUM_GRAPHS * F) return;
    sums[i] /= fmaxf(cnt[i / F], 1.f);
}
__global__ void mlp1(const float* __restrict__ g, const float* __restrict__ w,
                     const float* __restrict__ b, float* __restrict__ o) {
    int gi = blockIdx.x, f = threadIdx.x;   // 64 x 32
    const float* gp = g + gi * 512;
    const float* wp = w + f * 512;
    float s = b[f];
    for (int k = 0; k < 512; ++k) s += gp[k] * wp[k];
    o[gi * 32 + f] = s > 0.f ? s : (expf(s) - 1.f);
}
__global__ void mlp2(const float* __restrict__ g2, const float* __restrict__ w,
                     const float* __restrict__ b, float* __restrict__ out) {
    int gi = blockIdx.x, f = threadIdx.x;   // 64 x 2
    float s = b[f];
    for (int k = 0; k < 32; ++k) s += g2[gi * 32 + k] * w[f * 32 + k];
    out[gi * 2 + f] = s;
}

// ---------------------------------------------------------------------------
static inline size_t alignup(size_t x) { return (x + 255) & ~(size_t)255; }

extern "C" void kernel_launch(void* const* d_in, const int* in_sizes, int n_in,
                              void* d_out, int out_size, void* d_ws, size_t ws_size,
                              hipStream_t stream) {
    const int IN_DIM = 128;
    const int N  = in_sizes[0] / IN_DIM;        // 50000 (divisible by 16)
    const int E0 = in_sizes[1] / 2;             // 800000
    const int Et = E0 + N;                      // + self loops

    const float* x  = (const float*)d_in[0];
    const int* ei   = (const int*)d_in[1];
    const int* batch= (const int*)d_in[2];

    // workspace carve-out
    char* ws = (char*)d_ws;
    size_t off = 0;
    auto take = [&](size_t bytes) { char* p = ws + off; off += alignup(bytes); return p; };
    float*    act0  = (float*)take((size_t)N * 512 * 4);
    float*    act1  = (float*)take((size_t)N * 512 * 4);
    float*    hbuf  = (float*)take((size_t)N * 512 * 4);
    __bf16*   abf   = (__bf16*)take((size_t)N * 512 * 2);      // bf16 activations
    __bf16*   wbf   = (__bf16*)take((size_t)512 * 512 * 2);    // bf16 weights
    float*    es    = (float*)take((size_t)N * HEADS * 4);
    float*    ed    = (float*)take((size_t)N * HEADS * 4);
    unsigned* mu    = (unsigned*)take((size_t)N * HEADS * 4);
    float*    denom = (float*)take((size_t)N * HEADS * 4);
    float*    exv   = (float*)take((size_t)Et * HEADS * 4);
    int*      srcE  = (int*)take((size_t)Et * 4);
    int*      dstE  = (int*)take((size_t)Et * 4);
    float*    sums  = (float*)take((size_t)NUM_GRAPHS * 512 * 4);
    float*    cnt   = (float*)take((size_t)NUM_GRAPHS * 4);
    float*    g2    = (float*)take((size_t)NUM_GRAPHS * 32 * 4);
    (void)ws_size;

    build_edges<<<(Et + 255) / 256, 256, 0, stream>>>(ei, srcE, dstE, E0, N);

    struct Layer { int Fin, Fout, logC; const float *W, *as, *ad, *b; const float* in; float* agg; int elu; };
    Layer L[4] = {
        {128, 512, 6, (const float*)d_in[3],  (const float*)d_in[4],  (const float*)d_in[5],  (const float*)d_in[6],  x,    act0, 1},
        {512, 256, 5, (const float*)d_in[7],  (const float*)d_in[8],  (const float*)d_in[9],  (const float*)d_in[10], act0, act1, 1},
        {256, 128, 4, (const float*)d_in[11], (const float*)d_in[12], (const float*)d_in[13], (const float*)d_in[14], act1, act0, 1},
        {128, 512, 6, (const float*)d_in[15], (const float*)d_in[16], (const float*)d_in[17], (const float*)d_in[18], act0, act1, 0},
    };

    for (int l = 0; l < 4; ++l) {
        const Layer& P = L[l];
        const int C = P.Fout / HEADS;
        long nh = (long)N * HEADS;
        long na = (long)N * P.Fout;
        long ni = (long)N * P.Fin;
        long nw = (long)P.Fout * P.Fin;

        zero_f32<<<(int)((nh + 255) / 256), 256, 0, stream>>>((float*)mu, nh);
        zero_f32<<<(int)((nh + 255) / 256), 256, 0, stream>>>(denom, nh);
        zero_f32<<<(int)((na + 255) / 256), 256, 0, stream>>>(P.agg, na);

        // pre-cast GEMM operands to bf16 (keeps WMMA loop VALU-free)
        cast_bf16<<<(int)((ni / 2 + 255) / 256), 256, 0, stream>>>(P.in, abf, ni);
        cast_bf16<<<(int)((nw / 2 + 255) / 256), 256, 0, stream>>>(P.W, wbf, nw);

        dim3 gg(N / 16, P.Fout / 128);          // N % 16 == 0, Fout % 128 == 0
        gemm_wmma_bf16<<<gg, 64, 0, stream>>>(abf, wbf, hbuf, P.Fin, P.Fout);

        attn_coef<<<(int)((nh + 255) / 256), 256, 0, stream>>>(hbuf, P.as, P.ad, es, ed, N, C);

        long eh = (long)Et * HEADS;
        edge_max<<<(int)((eh + 255) / 256), 256, 0, stream>>>(srcE, dstE, es, ed, mu, Et);
        edge_exp<<<(int)((eh + 255) / 256), 256, 0, stream>>>(srcE, dstE, es, ed, mu, exv, denom, Et);
        edge_scatter<<<Et, P.Fout, 0, stream>>>(srcE, dstE, hbuf, exv, denom, P.agg, P.logC, P.Fout);

        bias_act<<<(int)((na + 255) / 256), 256, 0, stream>>>(P.agg, P.b, na, P.Fout, P.elu);
    }

    // mean pool over graphs + MLP head
    zero_f32<<<(NUM_GRAPHS * 512 + 255) / 256, 256, 0, stream>>>(sums, NUM_GRAPHS * 512);
    zero_f32<<<1, 256, 0, stream>>>(cnt, NUM_GRAPHS);
    long nf = (long)N * 512;
    pool_sum<<<(int)((nf + 255) / 256), 256, 0, stream>>>(act1, batch, sums, N, 512);
    pool_cnt<<<(N + 255) / 256, 256, 0, stream>>>(batch, cnt, N);
    pool_div<<<(NUM_GRAPHS * 512 + 255) / 256, 256, 0, stream>>>(sums, cnt, 512);

    mlp1<<<NUM_GRAPHS, 32, 0, stream>>>(sums, (const float*)d_in[19], (const float*)d_in[20], g2);
    mlp2<<<NUM_GRAPHS, 2, 0, stream>>>(g2, (const float*)d_in[21], (const float*)d_in[22], (float*)d_out);
}
